// SNN_63479616635536
// MI455X (gfx1250) — matrix operand order repaired and checked by reading, not compiled, not run on hardware
//
#include <hip/hip_runtime.h>

// ---------------------------------------------------------------------------
// SNN forward (norse LIF encoder -> LIF recurrent -> LI readout), MI455X.
// B=128, T=512, NIN=12, H=512, NOUT=4.
// 8 persistent workgroups (one per 16-row batch tile), 512 threads (16 waves).
// Recurrent GEMMs via v_wmma_f32_16x16x32_f16, f32 accumulation, fp16 weights
// streamed from L2; spike vectors exchanged through LDS.
// ---------------------------------------------------------------------------

typedef __attribute__((ext_vector_type(16))) _Float16 v16h;
typedef __attribute__((ext_vector_type(8)))  float    v8f;

#define HH   512
#define TT   512
#define NINP 12
#define BTL  16       // batch tile (WMMA M)
#define NB   128
#define NOUT 4
#define SDT  0.001f

union U16frag { v16h v; float4 f[2]; };

__device__ __forceinline__ float clamp01(float v) {
    return fminf(fmaxf(v, 0.0f), 1.0f);
}

// Convert the two 512x512 fp32 weight matrices to fp16 (row-major, unchanged
// layout: W[h_out][k] -> contiguous K, read as B fragments in this kernel's k = 16*half + s order).
__global__ void snn_w2h(const float* __restrict__ w_in_f,
                        const float* __restrict__ w_rec_f,
                        _Float16* __restrict__ w16) {
    int i = blockIdx.x * blockDim.x + threadIdx.x;
    if (i < HH * HH) {
        w16[i]           = (_Float16)w_in_f[i];
        w16[HH * HH + i] = (_Float16)w_rec_f[i];
    }
}

__global__ __launch_bounds__(512, 1)
void snn_run(const float* __restrict__ x,        // [B,T,NIN]
             const float* __restrict__ w_in,     // [H,NIN]
             const float* __restrict__ w_out,    // [NOUT,H]
             const _Float16* __restrict__ w16,   // [2][H][H] fp16 (in, rec)
             const float* __restrict__ ts_enc, const float* __restrict__ tm_enc,
             const float* __restrict__ ts_l1,  const float* __restrict__ tm_l1,
             const float* __restrict__ ts_out, const float* __restrict__ tm_out,
             float* __restrict__ out)            // [T,B,NOUT]
{
    // ---- LDS (≈57 KB) ----
    __shared__ _Float16 sh_zenc[BTL * HH] __attribute__((aligned(16)));   // z_enc, [b][h]
    __shared__ _Float16 sh_z1[2][BTL * HH] __attribute__((aligned(16)));  // z1 ping-pong
    __shared__ float    sh_xt[BTL * NINP];
    __shared__ float    sh_wout[NOUT * HH];
    __shared__ float    sh_outacc[BTL * NOUT];

    const int tid   = threadIdx.x;
    const int bt    = blockIdx.x;           // batch tile 0..7
    const int lane  = tid & 31;
    const int wv    = tid >> 5;             // wave 0..15
    const int col   = lane & 15;            // WMMA M (A) / N (B,C)
    const int hi    = lane >> 4;            // lane half
    const int mbase = hi * 8;               // C rows owned by this lane
    const int nt0   = wv * 2, nt1 = wv * 2 + 1;     // N tiles (h_out/16)
    const int h0    = nt0 * 16 + col;
    const int h1    = nt1 * 16 + col;

    const _Float16* __restrict__ wrec16 = w16 + HH * HH;

    // ---- one-time init ----
    for (int j = tid; j < NOUT * HH; j += 512) sh_wout[j] = w_out[j];
    for (int j = tid; j < BTL * HH;  j += 512) sh_z1[1][j] = (_Float16)0.0f;  // s1_z(0)=0
    if (tid < BTL * NOUT) sh_outacc[tid] = 0.0f;

    // Encoder constants: thread owns column h = tid for all 16 batch rows.
    float wk[NINP];
    #pragma unroll
    for (int k = 0; k < NINP; ++k) wk[k] = w_in[tid * NINP + k];
    const float aE = SDT * clamp01(tm_enc[tid]);
    const float bE = SDT * clamp01(ts_enc[tid]);
    // L1 constants in C-fragment layout (columns h0, h1).
    const float aL0 = SDT * clamp01(tm_l1[h0]);
    const float bL0 = SDT * clamp01(ts_l1[h0]);
    const float aL1 = SDT * clamp01(tm_l1[h1]);
    const float bL1 = SDT * clamp01(ts_l1[h1]);
    const float aO  = SDT * clamp01(tm_out[0]);
    const float bO  = SDT * clamp01(ts_out[0]);

    // States.
    float ev[BTL], ei[BTL];                 // encoder v,i  (h=tid, b=r)
    float lv[16], li[16], idec[16];         // l1 v,i in fragment layout
    #pragma unroll
    for (int r = 0; r < BTL; ++r) { ev[r] = 0.f; ei[r] = 0.f; }
    #pragma unroll
    for (int r = 0; r < 16; ++r) { lv[r] = 0.f; li[r] = 0.f; }
    float sov = 0.f, soi = 0.f;             // LI readout (threads < 64)

    __syncthreads();

    for (int t = 0; t < TT; ++t) {
        const int p = t & 1;

        // ---- phase A: stage x_t tile ----
        if (tid < BTL * NINP) {
            int r = tid / NINP, k = tid - r * NINP;
            sh_xt[tid] = x[((size_t)(bt * BTL + r) * TT + t) * NINP + k];
        }
        __syncthreads();

        // ---- phase B1: encoder LIF (elementwise, h = tid) ----
        #pragma unroll
        for (int r = 0; r < BTL; ++r) {
            float cur = 0.f;
            #pragma unroll
            for (int k = 0; k < NINP; ++k)
                cur = fmaf(sh_xt[r * NINP + k], wk[k], cur);
            float vd = ev[r] + aE * (ei[r] - ev[r]);
            float id = ei[r] * (1.0f - bE);
            float z  = (vd > 1.0f) ? 1.0f : 0.0f;     // spike(v_dec - VTH)
            ev[r] = (1.0f - z) * vd;
            ei[r] = id + cur;
            sh_zenc[r * HH + tid] = (_Float16)z;
        }
        // ---- phase B2: l1 LIF spike/decay (fragment layout) ----
        #pragma unroll
        for (int r = 0; r < 8; ++r) {
            {
                float vd = lv[r] + aL0 * (li[r] - lv[r]);
                idec[r]  = li[r] * (1.0f - bL0);
                float z  = (vd > 1.0f) ? 1.0f : 0.0f;
                lv[r] = (1.0f - z) * vd;
                sh_z1[p][(mbase + r) * HH + h0] = (_Float16)z;
            }
            {
                int q = 8 + r;
                float vd = lv[q] + aL1 * (li[q] - lv[q]);
                idec[q]  = li[q] * (1.0f - bL1);
                float z  = (vd > 1.0f) ? 1.0f : 0.0f;
                lv[q] = (1.0f - z) * vd;
                sh_z1[p][(mbase + r) * HH + h1] = (_Float16)z;
            }
        }
        __syncthreads();

        // ---- phase C1: readout GEMM partials (z1 @ w_out.T), VALU ----
        {
            int pi = tid >> 3, sub = tid & 7;       // 64 (b,o) pairs x 8 lanes
            int ob = pi >> 2, oo = pi & 3;
            const _Float16* zr = &sh_z1[p][ob * HH];
            const float*    wr = &sh_wout[oo * HH];
            float part = 0.f;
            #pragma unroll 8
            for (int hh = sub * 64; hh < sub * 64 + 64; ++hh)
                part = fmaf((float)zr[hh], wr[hh], part);
            atomicAdd(&sh_outacc[pi], part);        // ds_add_f32
        }

        // ---- phase C2: recurrent GEMMs via WMMA ----
        {
            v8f c0 = {}; v8f c1 = {};
            const _Float16* __restrict__ zprev = sh_z1[p ^ 1];
            #pragma unroll 4
            for (int kk = 0; kk < 16; ++kk) {
                // A fragments (16x32 f16) from LDS: two b128 per fragment.
                const int abase = col * HH + kk * 32 + hi * 16;
                U16frag aEn, aRe;
                aEn.f[0] = *(const float4*)(sh_zenc + abase);
                aEn.f[1] = *(const float4*)(sh_zenc + abase + 8);
                aRe.f[0] = *(const float4*)(zprev + abase);
                aRe.f[1] = *(const float4*)(zprev + abase + 8);
                // B fragments (32x16 f16) from L2-resident fp16 weights:
                // lane holds column h_out, contiguous K -> two b128.
                const int b0 = h0 * HH + kk * 32 + hi * 16;
                const int b1 = h1 * HH + kk * 32 + hi * 16;
                U16frag bi0, br0, bi1, br1;
                bi0.f[0] = *(const float4*)(w16 + b0);
                bi0.f[1] = *(const float4*)(w16 + b0 + 8);
                br0.f[0] = *(const float4*)(wrec16 + b0);
                br0.f[1] = *(const float4*)(wrec16 + b0 + 8);
                bi1.f[0] = *(const float4*)(w16 + b1);
                bi1.f[1] = *(const float4*)(w16 + b1 + 8);
                br1.f[0] = *(const float4*)(wrec16 + b1);
                br1.f[1] = *(const float4*)(wrec16 + b1 + 8);

                c0 = __builtin_amdgcn_wmma_f32_16x16x32_f16(
                         false, aEn.v, false, bi0.v, (short)0, c0, false, false);
                c0 = __builtin_amdgcn_wmma_f32_16x16x32_f16(
                         false, aRe.v, false, br0.v, (short)0, c0, false, false);
                c1 = __builtin_amdgcn_wmma_f32_16x16x32_f16(
                         false, aEn.v, false, bi1.v, (short)0, c1, false, false);
                c1 = __builtin_amdgcn_wmma_f32_16x16x32_f16(
                         false, aRe.v, false, br1.v, (short)0, c1, false, false);
            }
            // s1_i = i_dec + z_enc@Win.T + z1_prev@Wrec.T  (C layout == state layout)
            #pragma unroll
            for (int r = 0; r < 8; ++r) {
                li[r]     = idec[r]     + c0[r];
                li[8 + r] = idec[8 + r] + c1[r];
            }
        }
        __syncthreads();

        // ---- phase D: LI readout cell + emit voltage ----
        if (tid < BTL * NOUT) {
            float oin = sh_outacc[tid];
            sov = sov + aO * (soi - sov);           // uses old so_i
            soi = soi * (1.0f - bO) + oin;
            int b = tid >> 2, o = tid & 3;
            out[((size_t)t * NB + bt * BTL + b) * NOUT + o] = sov;
            sh_outacc[tid] = 0.0f;                  // re-zero for next step
        }
        // no trailing barrier needed: next write to any shared buffer is
        // separated from this step's readers by the phase-A/B barriers.
    }
}

extern "C" void kernel_launch(void* const* d_in, const int* in_sizes, int n_in,
                              void* d_out, int out_size, void* d_ws, size_t ws_size,
                              hipStream_t stream) {
    const float* x        = (const float*)d_in[0];
    const float* w_in     = (const float*)d_in[1];
    const float* w_l1_in  = (const float*)d_in[2];
    const float* w_l1_rec = (const float*)d_in[3];
    const float* w_out    = (const float*)d_in[4];
    const float* ts_enc   = (const float*)d_in[5];
    const float* tm_enc   = (const float*)d_in[6];
    const float* ts_l1    = (const float*)d_in[7];
    const float* tm_l1    = (const float*)d_in[8];
    const float* ts_out   = (const float*)d_in[9];
    const float* tm_out   = (const float*)d_in[10];

    _Float16* w16 = (_Float16*)d_ws;   // 2 * 512*512 fp16 = 1 MB scratch

    snn_w2h<<<(HH * HH + 255) / 256, 256, 0, stream>>>(w_l1_in, w_l1_rec, w16);

    snn_run<<<NB / BTL, 512, 0, stream>>>(
        x, w_in, w_out, w16,
        ts_enc, tm_enc, ts_l1, tm_l1, ts_out, tm_out,
        (float*)d_out);
}
